// Attention_62156766708152
// MI455X (gfx1250) — matrix operand order repaired and checked by reading, not compiled, not run on hardware
//
#include <hip/hip_runtime.h>
#include <hip/hip_bf16.h>

// ---------------- types / helpers -------------------------------------------
typedef __attribute__((ext_vector_type(16))) __bf16 v16bf;
typedef __attribute__((ext_vector_type(8)))  float  v8f;

struct BF32 { float4 lo, hi; };   // 32 bytes, matches v16bf

__device__ static inline unsigned short f2bf(float f) {
    unsigned u = __builtin_bit_cast(unsigned, f);
    unsigned r = u + 0x7FFFu + ((u >> 16) & 1u);   // round-to-nearest-even
    return (unsigned short)(r >> 16);
}

__device__ static inline v8f wmma_bf16(v16bf a, v16bf b, v8f c) {
    return __builtin_amdgcn_wmma_f32_16x16x32_bf16(false, a, false, b,
                                                   (short)0, c, false, false);
}

// A fragment (16x32 bf16): lane<16 -> row=lane, K chunks [0..7],[16..23];
// lane>=16 -> same row, K chunks [8..15],[24..31].
__device__ static inline v16bf frag_a(const unsigned short* base, int stride_us,
                                      int mrow, int koff, int lane) {
    const unsigned short* p = base + (size_t)(mrow + (lane & 15)) * stride_us
                                   + koff + ((lane & 16) ? 8 : 0);
    BF32 t;
    t.lo = *reinterpret_cast<const float4*>(p);
    t.hi = *reinterpret_cast<const float4*>(p + 16);
    return __builtin_bit_cast(v16bf, t);
}

// B fragment (32x16 bf16): lane<16 -> col=lane, K chunks [0..7],[16..23]; lane>=16 -> col=lane-16, K chunks [8..15],[24..31].
__device__ static inline v16bf frag_b(const unsigned short* base, int stride_us,
                                      int ncol, int koff, int lane) {
    const unsigned short* p = base + (size_t)(ncol + (lane & 15)) * stride_us
                                   + koff + ((lane & 16) ? 8 : 0);
    BF32 t;
    t.lo = *reinterpret_cast<const float4*>(p);
    t.hi = *reinterpret_cast<const float4*>(p + 16);
    return __builtin_bit_cast(v16bf, t);
}

// CDNA5 async memory->LDS copy (16B per lane), tracked by ASYNCcnt.
__device__ static inline unsigned lds_addr(const void* p) {
    return (unsigned)(size_t)p;   // generic LDS addr: low 32 bits = LDS offset
}
__device__ static inline void async_copy_b128(unsigned lds_off, const void* gsrc) {
    asm volatile("global_load_async_to_lds_b128 %0, %1, off"
                 :: "v"(lds_off), "v"(gsrc) : "memory");
}
__device__ static inline void wait_async_0()  { asm volatile("s_wait_asynccnt 0"  ::: "memory"); }
__device__ static inline void wait_async_8()  { asm volatile("s_wait_asynccnt 8"  ::: "memory"); }
__device__ static inline void wait_async_16() { asm volatile("s_wait_asynccnt 16" ::: "memory"); }

// ---------------- problem constants -----------------------------------------
#define B_  2
#define S_  2048
#define D_  4096
#define H_  32
#define HD_ 128
#define R_  16
#define M_  (B_ * S_)           // 4096 rows total

// ---------------- kernel 0: fp32 -> bf16 bulk conversion --------------------
__global__ void __launch_bounds__(256)
cvt_bf16(const float* __restrict__ src, unsigned short* __restrict__ dst) {
    size_t i = ((size_t)blockIdx.x * 256 + threadIdx.x) * 8;
    float4 a = *reinterpret_cast<const float4*>(src + i);
    float4 b = *reinterpret_cast<const float4*>(src + i + 4);
    union { unsigned short u[8]; uint4 v; } o;
    o.u[0] = f2bf(a.x); o.u[1] = f2bf(a.y); o.u[2] = f2bf(a.z); o.u[3] = f2bf(a.w);
    o.u[4] = f2bf(b.x); o.u[5] = f2bf(b.y); o.u[6] = f2bf(b.z); o.u[7] = f2bf(b.w);
    *reinterpret_cast<uint4*>(dst + i) = o.v;
}

// ---------------- kernel 1: LoRA down projections ---------------------------
__global__ void __launch_bounds__(256)
lora_down(const float* __restrict__ x,
          const float* __restrict__ lqd, const float* __restrict__ lvd,
          float* __restrict__ dq, float* __restrict__ dv) {
    int lane = threadIdx.x & 31, wave = threadIdx.x >> 5;
    int row  = blockIdx.x * 8 + wave;
    const float* xr = x + (size_t)row * D_;
    const float* w  = ((lane < 16) ? lqd : lvd) + (size_t)(lane & 15) * D_;
    float acc = 0.f;
    for (int d = 0; d < D_; d += 4) {
        float4 xv = *reinterpret_cast<const float4*>(xr + d);
        float4 wv = *reinterpret_cast<const float4*>(w + d);
        acc += xv.x * wv.x + xv.y * wv.y + xv.z * wv.z + xv.w * wv.w;
    }
    float* dst = ((lane < 16) ? dq : dv) + (size_t)row * R_ + (lane & 15);
    *dst = acc;
}

// ---------------- kernel 2/4: GEMM (bf16 WMMA, async double-buffered) -------
// mode 0: xq = x@wq^T + lora_q, RoPE  -> qb (bf16 [B,H,S,HD])
// mode 1: xk = x@wk^T,          RoPE  -> kb (bf16 [B,H,S,HD])
// mode 2: xv = x@wv^T + lora_v        -> vb (bf16 [B,H,HD,S]  TRANSPOSED)
// mode 3: out = attn(bf16)@wo^T       -> outF (fp32 [B,S,D])
#define GSTR 72   // ushorts per LDS row (64 data + 8 pad) = 144B, 16B aligned

__global__ void __launch_bounds__(256)
gemm_qkvo(const unsigned short* __restrict__ xbf, const unsigned short* __restrict__ abf,
          const unsigned short* __restrict__ wqb, const unsigned short* __restrict__ wkb,
          const unsigned short* __restrict__ wvb, const unsigned short* __restrict__ wob,
          const float* __restrict__ dq, const float* __restrict__ graph_rep,
          const float* __restrict__ dv, const float* __restrict__ lora_v_up,
          const float* __restrict__ fcos, const float* __restrict__ fsin,
          unsigned short* __restrict__ qb, unsigned short* __restrict__ kb,
          unsigned short* __restrict__ vb, float* __restrict__ outF,
          int modeBase) {
    int mode = modeBase + (int)blockIdx.z;
    const unsigned short* A = (mode < 3) ? xbf : abf;
    const unsigned short* W = (mode == 0) ? wqb : (mode == 1) ? wkb
                             : (mode == 2) ? wvb : wob;
    int nt = blockIdx.x, mt = blockIdx.y;
    int tid = threadIdx.x, lane = tid & 31, wave = tid >> 5;

    __shared__ union SMem {
        unsigned short buf[2][2][128 * GSTR];   // [parity][A=0/B=1]
        float C[128 * 132];
    } sm;

    const unsigned short* Arow = A + (size_t)(mt * 128) * D_;
    const unsigned short* Wrow = W + (size_t)(nt * 128) * D_;

    // per-thread: 4 A-chunks + 4 B-chunks (b128 = 8 bf16) per 64-K stage
    auto issue = [&](int parity, int kk) {
#pragma unroll
        for (int j = 0; j < 4; j++) {
            int id  = tid + 256 * j;
            int row = id >> 3, k8 = (id & 7) << 3;
            async_copy_b128(lds_addr(&sm.buf[parity][0][row * GSTR + k8]),
                            Arow + (size_t)row * D_ + kk + k8);
            async_copy_b128(lds_addr(&sm.buf[parity][1][row * GSTR + k8]),
                            Wrow + (size_t)row * D_ + kk + k8);
        }
    };

    v8f z = {0.f, 0.f, 0.f, 0.f, 0.f, 0.f, 0.f, 0.f};
    v8f acc[4][2];
#pragma unroll
    for (int i = 0; i < 4; i++)
#pragma unroll
        for (int j = 0; j < 2; j++) acc[i][j] = z;

    int wm = (wave >> 2) * 64;   // 2 wave rows x 64 M
    int wn = (wave & 3) * 32;    // 4 wave cols x 32 N

    issue(0, 0);
    for (int i = 0; i < 64; i++) {
        __syncthreads();                        // WAR on buf[(i+1)&1]
        if (i + 1 < 64) { issue((i + 1) & 1, (i + 1) * 64); wait_async_8(); }
        else            { wait_async_0(); }
        __syncthreads();                        // stage i visible to all
        const unsigned short* Ab = sm.buf[i & 1][0];
        const unsigned short* Bb = sm.buf[i & 1][1];
#pragma unroll
        for (int ks = 0; ks < 64; ks += 32) {
            v16bf b0 = frag_b(Bb, GSTR, wn,      ks, lane);
            v16bf b1 = frag_b(Bb, GSTR, wn + 16, ks, lane);
#pragma unroll
            for (int ii = 0; ii < 4; ii++) {
                v16bf a = frag_a(Ab, GSTR, wm + ii * 16, ks, lane);
                acc[ii][0] = wmma_bf16(a, b0, acc[ii][0]);
                acc[ii][1] = wmma_bf16(a, b1, acc[ii][1]);
            }
        }
    }

    __syncthreads();
    // dump accumulators -> LDS C tile (row-major, padded)
#pragma unroll
    for (int i = 0; i < 4; i++)
#pragma unroll
        for (int j = 0; j < 2; j++) {
            int r0 = wm + i * 16 + ((lane >> 4) << 3);
            int cN = wn + j * 16 + (lane & 15);
#pragma unroll
            for (int v = 0; v < 8; v++)
                sm.C[(r0 + v) * 132 + cN] = acc[i][j][v];
        }
    __syncthreads();

    const float lora_scale = 1.0f / (float)R_;
    if (mode == 3) {
        for (int idx = tid; idx < 128 * 128; idx += 256) {
            int r = idx >> 7, cN = idx & 127;
            int grow = mt * 128 + r;
            outF[(size_t)grow * D_ + nt * 128 + cN] = sm.C[r * 132 + cN];
        }
    } else {
        for (int idx = tid; idx < 128 * 64; idx += 256) {
            int r  = idx >> 6;
            int p  = idx & 63;
            int cN = p * 2;
            int grow = mt * 128 + r;          // = b*S + s
            int b    = grow >> 11;
            int sRow = grow & (S_ - 1);
            int e0   = nt * 128 + cN;
            float v0 = sm.C[r * 132 + cN];
            float v1 = sm.C[r * 132 + cN + 1];
            if (mode == 0) {
                const float* dqr = dq + (size_t)grow * R_;
                const float* g0  = graph_rep + (size_t)b * R_ * D_ + e0;
                float a0 = 0.f, a1 = 0.f;
#pragma unroll
                for (int rr = 0; rr < R_; rr++) {
                    float d = dqr[rr];
                    a0 += d * g0[(size_t)rr * D_];
                    a1 += d * g0[(size_t)rr * D_ + 1];
                }
                v0 += a0 * lora_scale; v1 += a1 * lora_scale;
            } else if (mode == 2) {
                const float* dvr = dv + (size_t)grow * R_;
                const float* u0  = lora_v_up + (size_t)e0 * R_;
                float a0 = 0.f, a1 = 0.f;
#pragma unroll
                for (int rr = 0; rr < R_; rr++) {
                    a0 += dvr[rr] * u0[rr];
                    a1 += dvr[rr] * u0[R_ + rr];
                }
                v0 += a0 * lora_scale; v1 += a1 * lora_scale;
            }
            int h  = nt;       // N tile spans exactly one head (HD==128)
            int hd = cN;
            if (mode <= 1) {   // RoPE on q and k
                float c  = fcos[sRow * (HD_ / 2) + p];
                float sn = fsin[sRow * (HD_ / 2) + p];
                float o0 = v0 * c - v1 * sn;
                float o1 = v0 * sn + v1 * c;
                unsigned short* dst = (mode == 0) ? qb : kb;
                size_t off = (((size_t)b * H_ + h) * S_ + sRow) * HD_ + hd;
                dst[off]     = f2bf(o0);
                dst[off + 1] = f2bf(o1);
            } else {           // V: transposed layout [B,H,HD,S]
                size_t off = (((size_t)b * H_ + h) * HD_ + hd) * S_ + sRow;
                vb[off]      = f2bf(v0);
                vb[off + S_] = f2bf(v1);
            }
        }
    }
}

// ---------------- kernel 3: flash attention ---------------------------------
#define FSTR 136   // ushorts per LDS row (128 data + 8 pad) = 272B, 16B aligned

__global__ void __launch_bounds__(256)
flash_attn(const unsigned short* __restrict__ qb,
           const unsigned short* __restrict__ kb,
           const unsigned short* __restrict__ vbT,   // [B,H,HD,S]
           unsigned short* __restrict__ ab /* bf16 [B,S,D] */) {
    int qt = blockIdx.x, h = blockIdx.y, b = blockIdx.z;
    int tid = threadIdx.x, lane = tid & 31, wave = tid >> 5;

    __shared__ unsigned short Qs[128 * FSTR];
    __shared__ unsigned short KV[2][2][128 * FSTR];   // [parity][K=0 / Vt=1]
    __shared__ unsigned short Ps[128 * FSTR];

    size_t headBase = ((size_t)b * H_ + h) * S_;
    const unsigned short* vhead = vbT + ((size_t)b * H_ + h) * (size_t)HD_ * S_;

    // async stage Q (8 chunks/thread)
#pragma unroll
    for (int j = 0; j < 8; j++) {
        int id = tid + 256 * j;
        int r = id >> 4, k8 = (id & 15) << 3;
        async_copy_b128(lds_addr(&Qs[r * FSTR + k8]),
                        qb + (headBase + qt * 128 + r) * HD_ + k8);
    }
    // async stage of one K tile (row-major) + one V tile (already transposed)
    auto issueKV = [&](int kt, int parity) {
#pragma unroll
        for (int j = 0; j < 8; j++) {
            int id = tid + 256 * j;
            int r = id >> 4, k8 = (id & 15) << 3;
            async_copy_b128(lds_addr(&KV[parity][0][r * FSTR + k8]),
                            kb + (headBase + kt * 128 + r) * HD_ + k8);
            async_copy_b128(lds_addr(&KV[parity][1][r * FSTR + k8]),
                            vhead + (size_t)r * S_ + kt * 128 + k8);
        }
    };
    issueKV(0, 0);

    v8f z = {0.f, 0.f, 0.f, 0.f, 0.f, 0.f, 0.f, 0.f};
    v8f oacc[8];
#pragma unroll
    for (int t = 0; t < 8; t++) oacc[t] = z;
    float mrow[8], lrow[8];
#pragma unroll
    for (int v = 0; v < 8; v++) { mrow[v] = -1.0e30f; lrow[v] = 0.f; }

    const float scale = 0.08838834764831845f;   // 1/sqrt(128)

    for (int kt = 0; kt <= qt; kt++) {
        __syncthreads();                              // WAR on KV[(kt+1)&1]
        if (kt + 1 <= qt) { issueKV(kt + 1, (kt + 1) & 1); wait_async_16(); }
        else              { wait_async_0(); }
        __syncthreads();                              // stage kt (and Q) visible
        const unsigned short* Ks = KV[kt & 1][0];
        const unsigned short* Vt = KV[kt & 1][1];

        // S = Q K^T : wave owns q-rows [wave*16, wave*16+16)
        v8f sacc[8];
#pragma unroll
        for (int t = 0; t < 8; t++) sacc[t] = z;
#pragma unroll
        for (int ks = 0; ks < 4; ks++) {
            v16bf a = frag_a(Qs, FSTR, wave * 16, ks * 32, lane);
#pragma unroll
            for (int t = 0; t < 8; t++) {
                v16bf bk = frag_b(Ks, FSTR, t * 16, ks * 32, lane);
                sacc[t] = wmma_bf16(a, bk, sacc[t]);
            }
        }

        // online softmax
        int rowg0 = qt * 128 + wave * 16 + ((lane >> 4) << 3);
        int colg0 = kt * 128 + (lane & 15);
#pragma unroll
        for (int vv = 0; vv < 8; vv++) {
            float rm = -1.0e30f;
#pragma unroll
            for (int t = 0; t < 8; t++) {
                float sv = sacc[t][vv] * scale;
                if (colg0 + t * 16 > rowg0 + vv) sv -= 1.0e9f;  // causal mask
                sacc[t][vv] = sv;
                rm = fmaxf(rm, sv);
            }
            for (int off = 1; off < 16; off <<= 1)
                rm = fmaxf(rm, __shfl_xor(rm, off, 32));
            float mn    = fmaxf(mrow[vv], rm);
            float alpha = __expf(mrow[vv] - mn);
            float rs = 0.f;
#pragma unroll
            for (int t = 0; t < 8; t++) {
                float p = __expf(sacc[t][vv] - mn);
                sacc[t][vv] = p;
                rs += p;
            }
            for (int off = 1; off < 16; off <<= 1)
                rs += __shfl_xor(rs, off, 32);
            lrow[vv] = lrow[vv] * alpha + rs;
            mrow[vv] = mn;
#pragma unroll
            for (int t = 0; t < 8; t++) oacc[t][vv] *= alpha;
        }

        // store P (bf16) into this wave's private LDS band
        int r0 = wave * 16 + ((lane >> 4) << 3);
#pragma unroll
        for (int t = 0; t < 8; t++) {
            int cc = t * 16 + (lane & 15);
#pragma unroll
            for (int vv = 0; vv < 8; vv++)
                Ps[(r0 + vv) * FSTR + cc] = f2bf(sacc[t][vv]);
        }
        asm volatile("s_wait_dscnt 0" ::: "memory");  // intra-wave LDS RAW

        // O += P * V
#pragma unroll
        for (int ks = 0; ks < 4; ks++) {
            v16bf a = frag_a(Ps, FSTR, wave * 16, ks * 32, lane);
#pragma unroll
            for (int t = 0; t < 8; t++) {
                v16bf bv = frag_b(Vt, FSTR, t * 16, ks * 32, lane);
                oacc[t] = wmma_bf16(a, bv, oacc[t]);
            }
        }
    }

    // finalize: O /= l ; write bf16 [B,S,D] with e = h*128+hd
    int r0 = wave * 16 + ((lane >> 4) << 3);
#pragma unroll
    for (int vv = 0; vv < 8; vv++) {
        float inv = 1.0f / lrow[vv];
        int sRow = qt * 128 + r0 + vv;
        size_t rowOff = ((size_t)b * S_ + sRow) * D_ + (size_t)h * HD_;
#pragma unroll
        for (int t = 0; t < 8; t++)
            ab[rowOff + t * 16 + (lane & 15)] = f2bf(oacc[t][vv] * inv);
    }
}

// ---------------- launch ----------------------------------------------------
extern "C" void kernel_launch(void* const* d_in, const int* in_sizes, int n_in,
                              void* d_out, int out_size, void* d_ws, size_t ws_size,
                              hipStream_t stream) {
    const float* x    = (const float*)d_in[0];
    const float* fcos = (const float*)d_in[2];
    const float* fsin = (const float*)d_in[3];
    const float* grep = (const float*)d_in[4];
    const float* wq   = (const float*)d_in[5];
    const float* wk   = (const float*)d_in[6];
    const float* wv   = (const float*)d_in[7];
    const float* wo   = (const float*)d_in[8];
    const float* lqd  = (const float*)d_in[9];
    const float* lvd  = (const float*)d_in[10];
    const float* lvu  = (const float*)d_in[11];
    float* out = (float*)d_out;

    // workspace layout (all tensors are 16,777,216 elements)
    size_t E = (size_t)16777216;
    unsigned short* xbf = (unsigned short*)d_ws;
    unsigned short* wqb = xbf + E;
    unsigned short* wkb = wqb + E;
    unsigned short* wvb = wkb + E;
    unsigned short* wob = wvb + E;
    unsigned short* qbuf = wob + E;
    unsigned short* kbuf = qbuf + E;
    unsigned short* vbuf = kbuf + E;   // [B,H,HD,S]
    unsigned short* abuf = vbuf + E;   // attention output, bf16 [B,S,D]
    float* dq = (float*)(abuf + E);
    float* dv = dq + (size_t)M_ * R_;

    // 0) bulk fp32 -> bf16 (x and the four weights)
    int cvtGrid = (int)(E / (256 * 8));   // 8192 blocks
    cvt_bf16<<<dim3(cvtGrid), 256, 0, stream>>>(x,  xbf);
    cvt_bf16<<<dim3(cvtGrid), 256, 0, stream>>>(wq, wqb);
    cvt_bf16<<<dim3(cvtGrid), 256, 0, stream>>>(wk, wkb);
    cvt_bf16<<<dim3(cvtGrid), 256, 0, stream>>>(wv, wvb);
    cvt_bf16<<<dim3(cvtGrid), 256, 0, stream>>>(wo, wob);

    // 1) LoRA down projections
    lora_down<<<dim3(M_ / 8), 256, 0, stream>>>(x, lqd, lvd, dq, dv);

    // 2) Q/K/V projections + LoRA-up + RoPE (modes 0..2 via blockIdx.z)
    gemm_qkvo<<<dim3(D_ / 128, M_ / 128, 3), 256, 0, stream>>>(
        xbf, abuf, wqb, wkb, wvb, wob, dq, grep, dv, lvu, fcos, fsin,
        qbuf, kbuf, vbuf, out, 0);

    // 3) causal flash attention
    flash_attn<<<dim3(S_ / 128, H_, B_), 256, 0, stream>>>(qbuf, kbuf, vbuf, abuf);

    // 4) output projection (mode 3)
    gemm_qkvo<<<dim3(D_ / 128, M_ / 128, 1), 256, 0, stream>>>(
        xbf, abuf, wqb, wkb, wvb, wob, dq, grep, dv, lvu, fcos, fsin,
        qbuf, kbuf, vbuf, out, 3);
}
